// MultiHeadAttention_61134564491984
// MI455X (gfx1250) — matrix operand order repaired and checked by reading, not compiled, not run on hardware
//
#include <hip/hip_runtime.h>
#include <hip/hip_bf16.h>

// ---------------------------------------------------------------------------
// MI455X (gfx1250) multi-head attention, bf16 WMMA pipeline.
//   x:[4,1024,1024] f32; Wq/Wk/Wv/Wo:[1024,1024] f32; bo:[1024] f32
//   out = MHA(x) @ Wo.T + bo   (reference applies NO causal mask)
// Working set (~56MB) fits in 192MB L2 -> feed WMMA fragments straight from
// global (L2), skip LDS GEMM tiling. All matmuls use v_wmma_f32_16x16x32_bf16.
// ---------------------------------------------------------------------------

typedef unsigned short u16;
typedef __bf16 bf16_t;
typedef bf16_t v16bf __attribute__((ext_vector_type(16)));
typedef float  v8f   __attribute__((ext_vector_type(8)));
typedef u16    v8u   __attribute__((ext_vector_type(8)));
typedef u16    v16u  __attribute__((ext_vector_type(16)));

#define DMODEL 1024
#define SEQ    1024
#define NHEAD  16
#define HDIM   64

static __device__ inline u16 f32_to_bf16_bits(float f) {
  unsigned x = __builtin_bit_cast(unsigned, f);
  unsigned r = x + 0x7FFFu + ((x >> 16) & 1u);   // round-to-nearest-even
  return (u16)(r >> 16);
}

static __device__ inline v16bf make_frag(v8u lo, v8u hi) {
  v16u t = __builtin_shufflevector(lo, hi, 0,1,2,3,4,5,6,7,8,9,10,11,12,13,14,15);
  return __builtin_bit_cast(v16bf, t);
}

// A fragment (16x32, bf16, row-major source, leading dim ld):
//   lane: row = row0 + (lane&15); kg = lane>>4
//   halves 0-7  <- K = k0 + 8*kg  .. +7
//   halves 8-15 <- K = k0 + 16 + 8*kg .. +7
static __device__ inline v16bf load_a_g(const u16* __restrict__ base, int row0,
                                        int ld, int k0, int lane) {
  int r  = row0 + (lane & 15);
  int kg = lane >> 4;
  const u16* p = base + (size_t)r * ld + k0 + 8 * kg;
  v8u lo = *(const v8u*)p;
  v8u hi = *(const v8u*)(p + 16);
  return make_frag(lo, hi);
}

// B fragment (32x16, bf16). B[k][n] = SRC[n][k] (SRC row-major, ld):
//   lane: n = col0 + (lane&15); kg = lane>>4; halves 0-7 <- K = k0+8*kg .. +7, halves 8-15 <- K = k0+16+8*kg .. +7
static __device__ inline v16bf load_b_g(const u16* __restrict__ base, int col0,
                                        int ld, int k0, int lane) {
  int n  = col0 + (lane & 15);
  int kg = lane >> 4;
  const u16* p = base + (size_t)n * ld + k0 + 8 * kg;
  v8u lo = *(const v8u*)p;
  v8u hi = *(const v8u*)(p + 16);
  return make_frag(lo, hi);
}

static __device__ inline v8f wmma_bf16(v16bf a, v16bf b, v8f c) {
  return __builtin_amdgcn_wmma_f32_16x16x32_bf16(false, a, false, b,
                                                 (short)0, c, false, false);
}

static __device__ inline v8f max8(v8f a, v8f b) {
  v8f r;
#pragma unroll
  for (int i = 0; i < 8; ++i) r[i] = fmaxf(a[i], b[i]);
  return r;
}
static __device__ inline v8f shfl_xor8(v8f a, int mask) {
  v8f r;
#pragma unroll
  for (int i = 0; i < 8; ++i) r[i] = __shfl_xor(a[i], mask, 32);
  return r;
}
static __device__ inline v8f exp8(v8f a) {
  v8f r;
#pragma unroll
  for (int i = 0; i < 8; ++i) r[i] = __expf(a[i]);
  return r;
}

// ---------------------------------------------------------------------------
// f32 -> bf16 cast (vectorized x4)
// ---------------------------------------------------------------------------
__global__ void cast_f32_bf16_kernel(const float* __restrict__ in,
                                     u16* __restrict__ out, int n4) {
  int i = blockIdx.x * blockDim.x + threadIdx.x;
  if (i >= n4) return;
  const float4 v = ((const float4*)in)[i];
  ushort4 o;
  o.x = f32_to_bf16_bits(v.x);
  o.y = f32_to_bf16_bits(v.y);
  o.z = f32_to_bf16_bits(v.z);
  o.w = f32_to_bf16_bits(v.w);
  ((ushort4*)out)[i] = o;
}

// ---------------------------------------------------------------------------
// Generic WMMA GEMM: C[m][n] = sum_k A[m][k] * B[n][k]   (i.e. A @ B.T)
// block = 128 threads (4 waves, 2x2), wave tile 32x64, block tile 64x128.
// MODE 0: C -> bf16 row-major [M][N]
// MODE 1: C -> bf16 scattered V-transpose: Vt[(b*16+h)*64+d][s]
// MODE 2: C -> f32 row-major + bias (final output)
// ---------------------------------------------------------------------------
template <int MODE>
__global__ __launch_bounds__(128) void gemm_wmma_kernel(
    const u16* __restrict__ A, const u16* __restrict__ B,
    u16* __restrict__ Cb, float* __restrict__ Cf,
    const float* __restrict__ bias, int M, int N, int K) {
  const int lane = threadIdx.x & 31;
  const int wave = threadIdx.x >> 5;
  const int m0 = blockIdx.y * 64 + (wave & 1) * 32;
  const int n0 = blockIdx.x * 128 + (wave >> 1) * 64;

  v8f acc[2][4] = {};

  for (int k0 = 0; k0 < K; k0 += 32) {
    v16bf a0 = load_a_g(A, m0, K, k0, lane);
    v16bf a1 = load_a_g(A, m0 + 16, K, k0, lane);
    v16bf b0 = load_b_g(B, n0, K, k0, lane);
    v16bf b1 = load_b_g(B, n0 + 16, K, k0, lane);
    v16bf b2 = load_b_g(B, n0 + 32, K, k0, lane);
    v16bf b3 = load_b_g(B, n0 + 48, K, k0, lane);
    acc[0][0] = wmma_bf16(a0, b0, acc[0][0]);
    acc[0][1] = wmma_bf16(a0, b1, acc[0][1]);
    acc[0][2] = wmma_bf16(a0, b2, acc[0][2]);
    acc[0][3] = wmma_bf16(a0, b3, acc[0][3]);
    acc[1][0] = wmma_bf16(a1, b0, acc[1][0]);
    acc[1][1] = wmma_bf16(a1, b1, acc[1][1]);
    acc[1][2] = wmma_bf16(a1, b2, acc[1][2]);
    acc[1][3] = wmma_bf16(a1, b3, acc[1][3]);
  }

  // C/D layout: vgpr r, lanes 0-15: row=r, n=lane; lanes 16-31: row=r+8, n=lane-16
  const int hi = lane >> 4, lo = lane & 15;
#pragma unroll
  for (int mi = 0; mi < 2; ++mi) {
#pragma unroll
    for (int nt = 0; nt < 4; ++nt) {
      const int n = n0 + 16 * nt + lo;
      float bn = 0.0f;
      if constexpr (MODE == 2) bn = bias[n];
#pragma unroll
      for (int r = 0; r < 8; ++r) {
        const int m = m0 + 16 * mi + r + 8 * hi;
        const float v = acc[mi][nt][r];
        if constexpr (MODE == 0) {
          Cb[(size_t)m * N + n] = f32_to_bf16_bits(v);
        } else if constexpr (MODE == 1) {
          const int b = m >> 10, s = m & 1023;
          const int h = n >> 6, d = n & 63;
          Cb[(size_t)(((b * NHEAD + h) * HDIM + d)) * SEQ + s] = f32_to_bf16_bits(v);
        } else {
          Cf[(size_t)m * N + n] = v + bn;
        }
      }
    }
  }
}

// ---------------------------------------------------------------------------
// Fused flash-style attention (no mask, per reference).
// grid = (64 bh, 8), block = 128 (4 waves). Each wave: 32 q-rows of one (b,h).
// Online softmax stats (m,l) kept in WMMA C layout (row = r + 8*(lane>=16)),
// so rescaling of accumulators is pure per-lane VALU. Row max/sum reductions
// are shfl_xor within 16-lane halves. P goes through LDS (C-layout -> A-frag).
// ---------------------------------------------------------------------------
__global__ __launch_bounds__(128) void attn_flash_kernel(
    const u16* __restrict__ Q, const u16* __restrict__ Kmat,
    const u16* __restrict__ Vt, u16* __restrict__ Ctx) {
  const int lane = threadIdx.x & 31;
  const int wave = threadIdx.x >> 5;
  const int bh = blockIdx.x;
  const int b = bh >> 4, h = bh & 15;
  const int q0 = (blockIdx.y * 4 + wave) * 32;

  __shared__ u16 Plds[4][32][40];  // 32x32 P tile per wave, +8 pad (16B-aligned rows)

  const u16* Qb = Q    + (size_t)(b * SEQ) * DMODEL + h * HDIM;
  const u16* Kb = Kmat + (size_t)(b * SEQ) * DMODEL + h * HDIM;
  const u16* Vb = Vt   + (size_t)bh * HDIM * SEQ;

  // Q fragments: 2 M-tiles x 2 K-steps over head_dim=64, loaded once.
  v16bf qf[2][2];
#pragma unroll
  for (int mi = 0; mi < 2; ++mi)
#pragma unroll
    for (int dt = 0; dt < 2; ++dt)
      qf[mi][dt] = load_a_g(Qb, q0 + 16 * mi, DMODEL, dt * 32, lane);

  v8f acc[2][4] = {};
  v8f mrun[2], lrun[2];
#pragma unroll
  for (int mi = 0; mi < 2; ++mi) {
#pragma unroll
    for (int i = 0; i < 8; ++i) { mrun[mi][i] = -1e30f; lrun[mi][i] = 0.0f; }
  }

  const int hi = lane >> 4, lo = lane & 15;
  const float rsqrt_hd = 0.125f;  // 1/sqrt(64)

  for (int kk = 0; kk < SEQ; kk += 32) {
    // K fragments: 2 key-col tiles x 2 d-steps
    v16bf kf[2][2];
#pragma unroll
    for (int j = 0; j < 2; ++j)
#pragma unroll
      for (int dt = 0; dt < 2; ++dt)
        kf[j][dt] = load_b_g(Kb, kk + 16 * j, DMODEL, dt * 32, lane);

#pragma unroll
    for (int mi = 0; mi < 2; ++mi) {
      v8f z = {};
      v8f s0 = wmma_bf16(qf[mi][0], kf[0][0], z);
      s0 = wmma_bf16(qf[mi][1], kf[0][1], s0);
      v8f s1 = wmma_bf16(qf[mi][0], kf[1][0], z);
      s1 = wmma_bf16(qf[mi][1], kf[1][1], s1);
      s0 *= rsqrt_hd;
      s1 *= rsqrt_hd;

      // per-row max across this 32-col chunk (reduce over 16-lane halves)
      v8f t = max8(s0, s1);
      t = max8(t, shfl_xor8(t, 1));
      t = max8(t, shfl_xor8(t, 2));
      t = max8(t, shfl_xor8(t, 4));
      t = max8(t, shfl_xor8(t, 8));
      v8f mnew = max8(mrun[mi], t);
      v8f alpha = exp8(mrun[mi] - mnew);
      v8f p0 = exp8(s0 - mnew);
      v8f p1 = exp8(s1 - mnew);
      v8f rs = p0 + p1;
      rs += shfl_xor8(rs, 1);
      rs += shfl_xor8(rs, 2);
      rs += shfl_xor8(rs, 4);
      rs += shfl_xor8(rs, 8);
      lrun[mi] = lrun[mi] * alpha + rs;
      mrun[mi] = mnew;
#pragma unroll
      for (int nt = 0; nt < 4; ++nt) acc[mi][nt] *= alpha;

      // P (C layout) -> LDS (row-major 32x32 bf16)
#pragma unroll
      for (int r = 0; r < 8; ++r) {
        const int row = 16 * mi + r + 8 * hi;
        Plds[wave][row][lo]      = f32_to_bf16_bits(p0[r]);
        Plds[wave][row][16 + lo] = f32_to_bf16_bits(p1[r]);
      }
    }

    // P as A fragments (LDS read; same wave -> DS ops in order, backend waits)
    v16bf pa[2];
#pragma unroll
    for (int mi = 0; mi < 2; ++mi) {
      const int row = 16 * mi + lo;
      const int kg = hi;
      v8u l8 = *(const v8u*)&Plds[wave][row][8 * kg];
      v8u h8 = *(const v8u*)&Plds[wave][row][16 + 8 * kg];
      pa[mi] = make_frag(l8, h8);
    }
    // V^T fragments: B[k=key][n=d] = Vt[d][key]
    v16bf vf[4];
#pragma unroll
    for (int nt = 0; nt < 4; ++nt)
      vf[nt] = load_b_g(Vb, 16 * nt, SEQ, kk, lane);

#pragma unroll
    for (int mi = 0; mi < 2; ++mi)
#pragma unroll
      for (int nt = 0; nt < 4; ++nt)
        acc[mi][nt] = wmma_bf16(pa[mi], vf[nt], acc[mi][nt]);
  }

  // normalize and store ctx (bf16, [b,s,h*hd] layout)
#pragma unroll
  for (int mi = 0; mi < 2; ++mi) {
    v8f inv;
#pragma unroll
    for (int i = 0; i < 8; ++i) inv[i] = 1.0f / lrun[mi][i];
#pragma unroll
    for (int nt = 0; nt < 4; ++nt) {
      const int d = 16 * nt + lo;
#pragma unroll
      for (int r = 0; r < 8; ++r) {
        const int q = q0 + 16 * mi + r + 8 * hi;
        Ctx[(size_t)(b * SEQ + q) * DMODEL + h * HDIM + d] =
            f32_to_bf16_bits(acc[mi][nt][r] * inv[r]);
      }
    }
  }
}

// ---------------------------------------------------------------------------
// Host-side launch
// ---------------------------------------------------------------------------
extern "C" void kernel_launch(void* const* d_in, const int* in_sizes, int n_in,
                              void* d_out, int out_size, void* d_ws, size_t ws_size,
                              hipStream_t stream) {
  const float* x  = (const float*)d_in[0];
  const float* Wq = (const float*)d_in[1];
  const float* Wk = (const float*)d_in[2];
  const float* Wv = (const float*)d_in[3];
  const float* Wo = (const float*)d_in[4];
  const float* bo = (const float*)d_in[5];

  const int M = 4 * SEQ;       // 4096 rows (b*s)
  const int N = DMODEL;        // 1024
  const int K = DMODEL;        // 1024

  // workspace layout (bytes)
  char* ws = (char*)d_ws;
  u16* xb  = (u16*)(ws);                        // 8 MB
  u16* wqb = (u16*)(ws + (8u << 20));           // 2 MB
  u16* wkb = (u16*)(ws + (10u << 20));          // 2 MB
  u16* wvb = (u16*)(ws + (12u << 20));          // 2 MB
  u16* wob = (u16*)(ws + (14u << 20));          // 2 MB
  u16* Qb  = (u16*)(ws + (16u << 20));          // 8 MB
  u16* Kb  = (u16*)(ws + (24u << 20));          // 8 MB
  u16* Vtb = (u16*)(ws + (32u << 20));          // 8 MB  (per-head transposed V)
  u16* Ctx = (u16*)(ws + (40u << 20));          // 8 MB
  float* out = (float*)d_out;

  // 1) casts f32 -> bf16
  {
    int n4 = (M * K) / 4;  // x: 4M elems
    cast_f32_bf16_kernel<<<(n4 + 255) / 256, 256, 0, stream>>>(x, xb, n4);
    n4 = (N * K) / 4;      // weights: 1M elems each
    cast_f32_bf16_kernel<<<(n4 + 255) / 256, 256, 0, stream>>>(Wq, wqb, n4);
    cast_f32_bf16_kernel<<<(n4 + 255) / 256, 256, 0, stream>>>(Wk, wkb, n4);
    cast_f32_bf16_kernel<<<(n4 + 255) / 256, 256, 0, stream>>>(Wv, wvb, n4);
    cast_f32_bf16_kernel<<<(n4 + 255) / 256, 256, 0, stream>>>(Wo, wob, n4);
  }

  // 2) projections: Q = x@Wq.T, K = x@Wk.T, Vt = transpose_per_head(x@Wv.T)
  dim3 ggrid(N / 128, M / 64);
  gemm_wmma_kernel<0><<<ggrid, 128, 0, stream>>>(xb, wqb, Qb, nullptr, nullptr, M, N, K);
  gemm_wmma_kernel<0><<<ggrid, 128, 0, stream>>>(xb, wkb, Kb, nullptr, nullptr, M, N, K);
  gemm_wmma_kernel<1><<<ggrid, 128, 0, stream>>>(xb, wvb, Vtb, nullptr, nullptr, M, N, K);

  // 3) fused attention -> Ctx
  dim3 agrid(4 * NHEAD, SEQ / (4 * 32));
  attn_flash_kernel<<<agrid, 128, 0, stream>>>(Qb, Kb, Vtb, Ctx);

  // 4) output projection: out = Ctx@Wo.T + bo (f32)
  gemm_wmma_kernel<2><<<ggrid, 128, 0, stream>>>(Ctx, wob, nullptr, out, bo, M, N, K);
}
